// DrugGCN_12472585027971
// MI455X (gfx1250) — matrix operand order repaired and checked by reading, not compiled, not run on hardware
//
#include <hip/hip_runtime.h>
#include <hip/hip_bf16.h>

typedef __attribute__((ext_vector_type(16))) __bf16 v16bf;
typedef __attribute__((ext_vector_type(8)))  float  v8f;

#define FDIM 128

// ---------------------------------------------------------------- degree ----
__global__ void k_init_deg(float* __restrict__ deg, int N) {
  int i = blockIdx.x * blockDim.x + threadIdx.x;
  if (i < N) deg[i] = 1.0f;                       // +1 self loop
}

__global__ void k_deg_scatter(const int* __restrict__ dst, float* __restrict__ deg, int E) {
  int e = blockIdx.x * blockDim.x + threadIdx.x;
  if (e < E) unsafeAtomicAdd(&deg[dst[e]], 1.0f); // global_atomic_add_f32, L2-resident
}

__global__ void k_deg_fin(const float* __restrict__ deg, float* __restrict__ dinv,
                          float* __restrict__ disq, int N) {
  int i = blockIdx.x * blockDim.x + threadIdx.x;
  if (i < N) { float d = deg[i]; dinv[i] = 1.0f / d; disq[i] = rsqrtf(d); }
}

__global__ void k_norm(const int* __restrict__ src, const int* __restrict__ dst,
                       const float* __restrict__ disq, float* __restrict__ norm_e, int E) {
  int e = blockIdx.x * blockDim.x + threadIdx.x;
  if (e < E) norm_e[e] = disq[src[e]] * disq[dst[e]];
}

// ----------------------------------------------------- W fragment packing ----
// Pre-swizzle the 128x128 weight matrix into CDNA5 B-fragment lane order so the
// GEMM can load each lane's 16 bf16 as one contiguous 32B chunk.
// B 32x16 bf16 layout: lane -> column N=(lane&15), element e -> K = half*8+(e&7)+(e>>3)*16.
// Fragment id = kb*8 + nt;  storage: wfrag[(frag*32 + lane)*16 + e].
__global__ void k_pack_w(const float* __restrict__ W, __bf16* __restrict__ wfrag) {
  int fl = blockIdx.x * blockDim.x + threadIdx.x;   // fragment-lane id, 0..1023
  if (fl >= 32 * 32) return;
  int frag = fl >> 5, lane = fl & 31;
  int kb = frag >> 3, nt = frag & 7;
  int half = lane >> 4, c = lane & 15;
  v16bf v;
#pragma unroll
  for (int e = 0; e < 16; ++e)
    v[e] = (__bf16)W[(kb * 32 + half * 8 + (e & 7) + ((e >> 3) << 4)) * FDIM + nt * 16 + c];
  *(v16bf*)(wfrag + (size_t)fl * 16) = v;
}

// ------------------------------------------------------------- WMMA GEMM ----
// h = (relu?)(in) @ W ; agg = h * deg_inv + bias   (self-loop init value for
// the subsequent edge scatter-add).
// Block = 256 threads = 8 waves; block owns 128 rows x 128 cols of output;
// each wave owns a 16-row strip (8 column tiles, 8 v8f accumulators).
// A fragments staged in LDS in fragment order (32B/lane contiguous);
// B fragments read straight from the pre-packed global wfrag (L2-resident).
__global__ void __launch_bounds__(256)
k_gemm_wmma(const float* __restrict__ in, const __bf16* __restrict__ wfrag,
            const float* __restrict__ bias, const float* __restrict__ dinv,
            float* __restrict__ h, float* __restrict__ agg, int N, int relu_in) {
  __shared__ __align__(32) __bf16 sAfrag[32 * 32 * 16];   // (wave*4+kb, lane, e): 32KB

  const int tid  = threadIdx.x;
  const int wave = tid >> 5;
  const int lane = tid & 31;
  const int half = lane >> 4;
  const int r    = lane & 15;
  const int rowBase = blockIdx.x * 128;

  // ---- Stage A fragments: 1024 fragment-lanes, 4 per thread.
  // A 16x32 bf16 layout: lane half h, row M=(lane&15);
  //   e=0..7  -> K = kb*32 + h*8 + e
  //   e=8..15 -> K = kb*32 + 16 + h*8 + (e-8)
  // i.e. two contiguous 8-float runs -> 4 float4 global loads, 1 v16bf LDS store.
#pragma unroll
  for (int i = 0; i < 4; ++i) {
    int fl   = i * 256 + tid;
    int frag = fl >> 5;            // w*4 + kb
    int fln  = fl & 31;
    int w  = frag >> 2, kb = frag & 3;
    int fh = fln >> 4,  fr = fln & 15;
    int grow = rowBase + w * 16 + fr;
    float4 v0 = make_float4(0.f,0.f,0.f,0.f), v1 = v0, v2 = v0, v3 = v0;
    if (grow < N) {
      const float* p = in + (size_t)grow * FDIM + kb * 32 + fh * 8;
      v0 = *(const float4*)(p + 0);
      v1 = *(const float4*)(p + 4);
      v2 = *(const float4*)(p + 16);
      v3 = *(const float4*)(p + 20);
    }
    if (relu_in) {
      v0.x=fmaxf(v0.x,0.f); v0.y=fmaxf(v0.y,0.f); v0.z=fmaxf(v0.z,0.f); v0.w=fmaxf(v0.w,0.f);
      v1.x=fmaxf(v1.x,0.f); v1.y=fmaxf(v1.y,0.f); v1.z=fmaxf(v1.z,0.f); v1.w=fmaxf(v1.w,0.f);
      v2.x=fmaxf(v2.x,0.f); v2.y=fmaxf(v2.y,0.f); v2.z=fmaxf(v2.z,0.f); v2.w=fmaxf(v2.w,0.f);
      v3.x=fmaxf(v3.x,0.f); v3.y=fmaxf(v3.y,0.f); v3.z=fmaxf(v3.z,0.f); v3.w=fmaxf(v3.w,0.f);
    }
    v16bf a;
    a[0]=(__bf16)v0.x;  a[1]=(__bf16)v0.y;  a[2]=(__bf16)v0.z;  a[3]=(__bf16)v0.w;
    a[4]=(__bf16)v1.x;  a[5]=(__bf16)v1.y;  a[6]=(__bf16)v1.z;  a[7]=(__bf16)v1.w;
    a[8]=(__bf16)v2.x;  a[9]=(__bf16)v2.y;  a[10]=(__bf16)v2.z; a[11]=(__bf16)v2.w;
    a[12]=(__bf16)v3.x; a[13]=(__bf16)v3.y; a[14]=(__bf16)v3.z; a[15]=(__bf16)v3.w;
    *(v16bf*)(sAfrag + (size_t)fl * 16) = a;
  }
  __syncthreads();

  v8f acc[8];
#pragma unroll
  for (int nt = 0; nt < 8; ++nt) acc[nt] = (v8f){0.f,0.f,0.f,0.f,0.f,0.f,0.f,0.f};

#pragma unroll
  for (int kb = 0; kb < 4; ++kb) {          // K = 128 in blocks of 32
    v16bf a = *(const v16bf*)(sAfrag + (size_t)(((wave << 2) + kb) * 32 + lane) * 16);
#pragma unroll
    for (int nt = 0; nt < 8; ++nt) {
      v16bf b = *(const v16bf*)(wfrag + (size_t)(((kb << 3) + nt) * 32 + lane) * 16);
      acc[nt] = __builtin_amdgcn_wmma_f32_16x16x32_bf16(
          false, a, false, b, (short)0, acc[nt], false, false);
    }
  }

  // C/D layout: VGPR i, lanes 0-15 -> M=i, lanes 16-31 -> M=8+i; N = lane&15.
#pragma unroll
  for (int i = 0; i < 8; ++i) {
    int grow = rowBase + (wave << 4) + (half << 3) + i;
    if (grow < N) {
      float di = dinv[grow];
#pragma unroll
      for (int nt = 0; nt < 8; ++nt) {
        int col = (nt << 4) + r;
        float hv = acc[nt][i];
        h[(size_t)grow * FDIM + col]   = hv;
        agg[(size_t)grow * FDIM + col] = hv * di + bias[col];
      }
    }
  }
}

// ---------------------------------------------------------- edge scatter ----
// One wave per edge: lane handles 4 features (float4 gather, 4 atomic adds).
__global__ void k_edge_scatter(const float* __restrict__ h, const float* __restrict__ norm_e,
                               const int* __restrict__ src, const int* __restrict__ dst,
                               float* __restrict__ agg, int E) {
  int e    = (blockIdx.x * blockDim.x + threadIdx.x) >> 5;
  int lane = threadIdx.x & 31;
  if (e >= E) return;
  int s = src[e], d = dst[e];
  float w = norm_e[e];
  float4 v = *(const float4*)(h + (size_t)s * FDIM + (lane << 2));
  float* out = agg + (size_t)d * FDIM + (lane << 2);
  unsafeAtomicAdd(out + 0, v.x * w);
  unsafeAtomicAdd(out + 1, v.y * w);
  unsafeAtomicAdd(out + 2, v.z * w);
  unsafeAtomicAdd(out + 3, v.w * w);
}

// ---------------------------------------------------------------- pooling ----
__global__ void k_pool_init(float* __restrict__ sums, unsigned* __restrict__ maxu,
                            float* __restrict__ cnt, int G) {
  int i = blockIdx.x * blockDim.x + threadIdx.x;
  if (i < G * FDIM) { sums[i] = 0.f; maxu[i] = 0u; }  // 0u == bits of 0.0f (post-ReLU floor)
  if (i < G) cnt[i] = 0.f;
}

// One wave per node; fused final ReLU.
__global__ void k_pool_scatter(const float* __restrict__ agg, const int* __restrict__ gidx,
                               float* __restrict__ sums, unsigned* __restrict__ maxu,
                               float* __restrict__ cnt, int N) {
  int n    = (blockIdx.x * blockDim.x + threadIdx.x) >> 5;
  int lane = threadIdx.x & 31;
  if (n >= N) return;
  int g = gidx[n];
  float4 v = *(const float4*)(agg + (size_t)n * FDIM + (lane << 2));
  v.x = fmaxf(v.x, 0.f); v.y = fmaxf(v.y, 0.f);
  v.z = fmaxf(v.z, 0.f); v.w = fmaxf(v.w, 0.f);
  float*    sp = sums + (size_t)g * FDIM + (lane << 2);
  unsigned* mp = maxu + (size_t)g * FDIM + (lane << 2);
  unsafeAtomicAdd(sp + 0, v.x);  atomicMax(mp + 0, __float_as_uint(v.x));
  unsafeAtomicAdd(sp + 1, v.y);  atomicMax(mp + 1, __float_as_uint(v.y));
  unsafeAtomicAdd(sp + 2, v.z);  atomicMax(mp + 2, __float_as_uint(v.z));
  unsafeAtomicAdd(sp + 3, v.w);  atomicMax(mp + 3, __float_as_uint(v.w));
  if (lane == 0) unsafeAtomicAdd(cnt + g, 1.0f);
}

__global__ void k_pool_final(const float* __restrict__ sums, const unsigned* __restrict__ maxu,
                             const float* __restrict__ cnt, float* __restrict__ out, int G) {
  int i = blockIdx.x * blockDim.x + threadIdx.x;
  if (i >= G * 2 * FDIM) return;
  int g = i >> 8;
  int j = i & 255;
  out[i] = (j < FDIM) ? sums[(size_t)g * FDIM + j] / fmaxf(cnt[g], 1.0f)
                      : __uint_as_float(maxu[(size_t)g * FDIM + (j - FDIM)]);
}

// ------------------------------------------------------------------ launch ---
extern "C" void kernel_launch(void* const* d_in, const int* in_sizes, int n_in,
                              void* d_out, int out_size, void* d_ws, size_t ws_size,
                              hipStream_t stream) {
  const float* x    = (const float*)d_in[0];
  const int*   ei   = (const int*)d_in[1];
  const int*   gidx = (const int*)d_in[2];
  const float* W1   = (const float*)d_in[3];
  const float* b1   = (const float*)d_in[4];
  const float* W2   = (const float*)d_in[5];
  const float* b2   = (const float*)d_in[6];
  const float* W3   = (const float*)d_in[7];
  const float* b3   = (const float*)d_in[8];
  float* out = (float*)d_out;

  const int N = in_sizes[0] / FDIM;
  const int E = in_sizes[1] / 2;
  const int G = out_size / (2 * FDIM);
  const int* src = ei;
  const int* dst = ei + E;

  // workspace carve-up
  float* buf0 = (float*)d_ws;              // N*128
  float* buf1 = buf0 + (size_t)N * FDIM;   // N*128
  float* buf2 = buf1 + (size_t)N * FDIM;   // N*128
  float* deg  = buf2 + (size_t)N * FDIM;   // N
  float* dinv = deg  + N;                  // N
  float* disq = dinv + N;                  // N
  float* nrm  = disq + N;                  // E
  float* sums = nrm  + E;                  // G*128
  unsigned* maxu = (unsigned*)(sums + (size_t)G * FDIM); // G*128
  float* cnt = (float*)(maxu + (size_t)G * FDIM);        // G
  // 32B-aligned bf16 fragment buffers, 32KB each
  uintptr_t wp = ((uintptr_t)(cnt + G) + 63) & ~(uintptr_t)63;
  __bf16* wf1 = (__bf16*)wp;
  __bf16* wf2 = wf1 + 1024 * 16;
  __bf16* wf3 = wf2 + 1024 * 16;

  const int B = 256;
  auto cdiv = [](int a, int b) { return (a + b - 1) / b; };

  // normalization terms + weight fragment packing
  k_init_deg   <<<cdiv(N, B), B, 0, stream>>>(deg, N);
  k_deg_scatter<<<cdiv(E, B), B, 0, stream>>>(dst, deg, E);
  k_deg_fin    <<<cdiv(N, B), B, 0, stream>>>(deg, dinv, disq, N);
  k_norm       <<<cdiv(E, B), B, 0, stream>>>(src, dst, disq, nrm, E);
  k_pool_init  <<<cdiv(G * FDIM, B), B, 0, stream>>>(sums, maxu, cnt, G);
  k_pack_w     <<<4, B, 0, stream>>>(W1, wf1);
  k_pack_w     <<<4, B, 0, stream>>>(W2, wf2);
  k_pack_w     <<<4, B, 0, stream>>>(W3, wf3);

  const int gemmGrid = cdiv(N, 128);
  const int edgeGrid = cdiv(E * 32, B);

  // layer 1: in=x            h=buf0  agg=buf1
  k_gemm_wmma   <<<gemmGrid, B, 0, stream>>>(x,    wf1, b1, dinv, buf0, buf1, N, 0);
  k_edge_scatter<<<edgeGrid, B, 0, stream>>>(buf0, nrm, src, dst, buf1, E);
  // layer 2: in=relu(buf1)   h=buf2  agg=buf0
  k_gemm_wmma   <<<gemmGrid, B, 0, stream>>>(buf1, wf2, b2, dinv, buf2, buf0, N, 1);
  k_edge_scatter<<<edgeGrid, B, 0, stream>>>(buf2, nrm, src, dst, buf0, E);
  // layer 3: in=relu(buf0)   h=buf1  agg=buf2
  k_gemm_wmma   <<<gemmGrid, B, 0, stream>>>(buf0, wf3, b3, dinv, buf1, buf2, N, 1);
  k_edge_scatter<<<edgeGrid, B, 0, stream>>>(buf1, nrm, src, dst, buf2, E);

  // pooling (fused final ReLU)
  k_pool_scatter<<<cdiv(N * 32, B), B, 0, stream>>>(buf2, gidx, sums, maxu, cnt, N);
  k_pool_final  <<<cdiv(G * 2 * FDIM, B), B, 0, stream>>>(sums, maxu, cnt, out, G);
}
